// LSTM_65601330479593
// MI455X (gfx1250) — matrix operand order repaired and checked by reading, not compiled, not run on hardware
//
#include <hip/hip_runtime.h>
#include <stdint.h>

#define S_   512
#define B_   64
#define D_   512
#define H_   512
#define G4_  2048
#define NSB_ (S_ * B_)   // 32768 rows for the batched input GEMM
#define NWGR_ 16         // workgroups in the persistent recurrent kernel
#define LN_EPS 1e-5f

typedef __attribute__((ext_vector_type(16))) __bf16        v16bf;
typedef __attribute__((ext_vector_type(8)))  float         v8f;
typedef __attribute__((ext_vector_type(4)))  unsigned int  u32x4;

union Frag { v16bf v; u32x4 q[2]; };

static __device__ __forceinline__ unsigned short f2b(float f) {
  unsigned u = __builtin_bit_cast(unsigned, f);
  unsigned r = (u + 0x7fffu + ((u >> 16) & 1u)) >> 16;   // round-to-nearest-even
  return (unsigned short)r;
}
static __device__ __forceinline__ float b2f(unsigned short h) {
  unsigned u = ((unsigned)h) << 16;
  return __builtin_bit_cast(float, u);
}
static __device__ __forceinline__ float sigm(float x) { return 1.0f / (1.0f + __expf(-x)); }

static __device__ __forceinline__ v8f vzero() {
  v8f z;
#pragma unroll
  for (int i = 0; i < 8; ++i) z[i] = 0.0f;
  return z;
}

static __device__ __forceinline__ v8f wmma_bf16(v16bf a, v16bf b, v8f c) {
  return __builtin_amdgcn_wmma_f32_16x16x32_bf16(false, a, false, b, (short)0, c, false, false);
}

// Low 32 bits of a generic pointer to __shared__ == LDS byte offset (ISA aperture rule).
static __device__ __forceinline__ unsigned lds_off(const void* p) {
  return (unsigned)(unsigned long long)p;
}

// One 16B async global->LDS transfer per lane (GVS mode: SGPR base + 32-bit VGPR offset).
static __device__ __forceinline__ void async_copy16(unsigned ldsoff, unsigned goff, const void* base) {
  unsigned long long b = (unsigned long long)base;
  asm volatile("global_load_async_to_lds_b128 %0, %1, %2"
               :: "v"(ldsoff), "v"(goff), "s"(b) : "memory");
}
static __device__ __forceinline__ void wait_async0() {
  asm volatile("s_wait_asynccnt 0" ::: "memory");
}

// ---------------------------------------------------------------------------
// Pack wx/wh (f32 [L,512,2048]) into bf16 WMMA-B fragment order:
//   dest[((l*128+nt)*16+kc)*32+lane][e] = w[l][kc*32 + 16*(lane>>4) + e][nt*16 + (lane&15)]
// so each lane's 16 bf16 B-operand values are 32 contiguous bytes.
// ---------------------------------------------------------------------------
__global__ void k_pack(const float* __restrict__ wx, const float* __restrict__ wh,
                       unsigned short* __restrict__ wxp, unsigned short* __restrict__ whp) {
  unsigned idx = blockIdx.x * 256u + threadIdx.x;     // 2 * 512 * 2048 = 2,097,152 total
  unsigned e    = idx & 15u;
  unsigned lane = (idx >> 4) & 31u;
  unsigned kc   = (idx >> 9) & 15u;
  unsigned nt   = (idx >> 13) & 127u;
  unsigned l    = idx >> 20;
  unsigned k = kc * 32u + ((lane >> 4) << 4) + e;
  unsigned n = (nt << 4) + (lane & 15u);
  size_t src = ((size_t)l * D_ + k) * G4_ + n;
  wxp[idx] = f2b(wx[src]);
  whp[idx] = f2b(wh[src]);
}

__global__ void k_cvt(const float* __restrict__ in, unsigned short* __restrict__ out, int n) {
  int i = blockIdx.x * 256 + threadIdx.x;
  if (i < n) out[i] = f2b(in[i]);
}

__global__ void k_initH(const float* __restrict__ h0l, unsigned short* __restrict__ hb) {
  int i = blockIdx.x * 256 + threadIdx.x;
  if (i < B_ * H_) hb[i] = f2b(h0l[i]);
}

// ---------------------------------------------------------------------------
// Phase 1: gx = LayerNorm(x @ wx[l]) for all (s,b) rows, written as bf16.
// One WG per 16 rows; 8 waves x 16 N-tiles each; two passes (stats, normalize)
// so the full 2048-wide row never needs a raw f32 spill buffer.
// ---------------------------------------------------------------------------
__global__ void __launch_bounds__(256) k_p1(const unsigned short* __restrict__ xin,
                                            const unsigned short* __restrict__ wxpl,
                                            const float* __restrict__ lng,
                                            const float* __restrict__ lnb,
                                            unsigned short* __restrict__ gxout) {
  __shared__ __align__(16) unsigned short shA[16 * D_];   // 16 KB A tile
  __shared__ float shStat[8][16][2];
  __shared__ float shMu[16];
  __shared__ float shRs[16];

  const int tid = threadIdx.x;
  const int wave = tid >> 5, lane = tid & 31;
  const int nlo = lane & 15, hs = lane >> 4;
  const int blk = blockIdx.x;

  // Async-stage the 16x512 bf16 A tile into LDS.
  {
    const unsigned short* abase = xin + (size_t)blk * 16 * D_;
    for (int u = tid; u < 1024; u += 256)
      async_copy16(lds_off(shA) + (unsigned)u * 16u, (unsigned)u * 16u, abase);
    wait_async0();
    __syncthreads();
  }

  float rsum[8], rsq[8];
#pragma unroll
  for (int r = 0; r < 8; ++r) { rsum[r] = 0.f; rsq[r] = 0.f; }

  // ---- pass A: accumulate per-row sum / sumsq ----
  for (int t = 0; t < 16; ++t) {
    const int nt = wave * 16 + t;
    v8f acc = vzero();
    for (int kc = 0; kc < 16; ++kc) {
      Frag a, b;
      const char* ap = (const char*)shA + (size_t)((nlo * D_ + kc * 32 + hs * 16) * 2);
      a.q[0] = *(const u32x4*)ap;
      a.q[1] = *(const u32x4*)(ap + 16);
      const char* bp = (const char*)wxpl + ((size_t)(nt * 16 + kc) * 1024 + lane * 32);
      b.q[0] = ((const u32x4*)bp)[0];
      b.q[1] = ((const u32x4*)bp)[1];
      acc = wmma_bf16(a.v, b.v, acc);
    }
#pragma unroll
    for (int r = 0; r < 8; ++r) { float v = acc[r]; rsum[r] += v; rsq[r] += v * v; }
  }
  // reduce across the 16 columns of this half-wave (rows stay within half)
#pragma unroll
  for (int r = 0; r < 8; ++r)
    for (int mk = 1; mk < 16; mk <<= 1) {
      rsum[r] += __shfl_xor(rsum[r], mk, 32);
      rsq[r]  += __shfl_xor(rsq[r],  mk, 32);
    }
  if (nlo == 0) {
#pragma unroll
    for (int r = 0; r < 8; ++r) {
      shStat[wave][hs * 8 + r][0] = rsum[r];
      shStat[wave][hs * 8 + r][1] = rsq[r];
    }
  }
  __syncthreads();
  if (tid < 16) {
    float s = 0.f, q = 0.f;
    for (int wv = 0; wv < 8; ++wv) { s += shStat[wv][tid][0]; q += shStat[wv][tid][1]; }
    float mu  = s * (1.0f / (float)G4_);
    float var = q * (1.0f / (float)G4_) - mu * mu;
    shMu[tid] = mu;
    shRs[tid] = rsqrtf(var + LN_EPS);
  }
  __syncthreads();

  // ---- pass B: recompute, normalize, store bf16 ----
  for (int t = 0; t < 16; ++t) {
    const int nt = wave * 16 + t;
    v8f acc = vzero();
    for (int kc = 0; kc < 16; ++kc) {
      Frag a, b;
      const char* ap = (const char*)shA + (size_t)((nlo * D_ + kc * 32 + hs * 16) * 2);
      a.q[0] = *(const u32x4*)ap;
      a.q[1] = *(const u32x4*)(ap + 16);
      const char* bp = (const char*)wxpl + ((size_t)(nt * 16 + kc) * 1024 + lane * 32);
      b.q[0] = ((const u32x4*)bp)[0];
      b.q[1] = ((const u32x4*)bp)[1];
      acc = wmma_bf16(a.v, b.v, acc);
    }
    const int col = nt * 16 + nlo;
    const float g = lng[col], bb = lnb[col];
#pragma unroll
    for (int r = 0; r < 8; ++r) {
      const int lr = hs * 8 + r;
      const float v = (acc[r] - shMu[lr]) * shRs[lr] * g + bb;
      gxout[(size_t)(blk * 16 + lr) * G4_ + col] = f2b(v);
    }
  }
}

// ---------------------------------------------------------------------------
// Persistent recurrent kernel: 16 WGs x 256 threads.
// WG `wg` owns hidden columns [wg*32, wg*32+32); wave (m,p) computes the
// 16x16 tiles of all four gates for rows m*16..+15, column-half p.
// c-state lives in VGPRs for all 512 steps; wh slice lives in LDS (128 KB).
// ---------------------------------------------------------------------------
__device__ __forceinline__ void gridbar(unsigned* bar, unsigned& bexp) {
  bexp += 1;
  __syncthreads();
  if (threadIdx.x == 0) {
    __threadfence();
    if (atomicAdd(&bar[0], 1u) == NWGR_ - 1u) {
      atomicExch(&bar[0], 0u);
      __threadfence();
      atomicAdd(&bar[1], 1u);
    }
    while (atomicAdd(&bar[1], 0u) < bexp) __builtin_amdgcn_s_sleep(2);
  }
  __builtin_amdgcn_s_cluster_barrier();  // architectural NOP outside cluster dispatch
  __syncthreads();
  __threadfence();
}

__global__ void __launch_bounds__(256) k_rec(
    const unsigned short* __restrict__ gx, const unsigned short* __restrict__ whpl,
    const float* __restrict__ biasl, const float* __restrict__ lngh,
    const float* __restrict__ lnbh, const float* __restrict__ c0l,
    unsigned short* __restrict__ hbuf, float* __restrict__ gstats,
    unsigned* __restrict__ bar, float* __restrict__ outx,
    unsigned short* __restrict__ x1b, float* __restrict__ outh,
    float* __restrict__ outc, int is_last) {
  extern __shared__ char smem[];
  unsigned short* shB  = (unsigned short*)smem;                 // 131072 B: [8 ntile][16 kc][32 lane][32B]
  float*          shSt = (float*)(smem + 131072);               // [4 m][2 p][16 row][2]
  float*          shMu = (float*)(smem + 131072 + 1024);        // 64
  float*          shRs = shMu + 64;                             // 64

  const int tid = threadIdx.x, wg = blockIdx.x;
  const int wave = tid >> 5, lane = tid & 31;
  const int m = wave >> 1, p = wave & 1;
  const int nlo = lane & 15, hs = lane >> 4;
  const int row_a = m * 16 + nlo;                 // A-operand row (M = lane&15)
  const int kcol = wg * 32 + p * 16 + nlo;        // hidden column owned by this lane

  // Async-stage this WG's wh fragment slice (8 N-tiles x 16 K-chunks) into LDS.
  for (int i = tid; i < 8192; i += 256) {
    const int unit = i & 63, kc = (i >> 6) & 15, j = i >> 10;
    const int nt = (j >> 1) * 32 + wg * 2 + (j & 1);
    async_copy16(lds_off(shB) + (unsigned)i * 16u,
                 (unsigned)((nt * 16 + kc) * 1024 + unit * 16), whpl);
  }
  wait_async0();
  __syncthreads();

  float bia[4], gg[4], gb[4];
  int colg[4];
#pragma unroll
  for (int g = 0; g < 4; ++g) {
    colg[g] = g * H_ + kcol;
    bia[g] = biasl[colg[g]];
    gg[g]  = lngh[colg[g]];
    gb[g]  = lnbh[colg[g]];
  }
  int crow[8];
  float cst[8], hfin[8];
#pragma unroll
  for (int r = 0; r < 8; ++r) {
    crow[r] = m * 16 + hs * 8 + r;                // C/D row of accumulator element r
    cst[r]  = c0l[crow[r] * H_ + kcol];
    hfin[r] = 0.f;
  }

  unsigned bexp = 0;

  for (int s = 0; s < S_; ++s) {
    const unsigned short* hsrc = hbuf + (size_t)(s & 1) * (B_ * H_);
    unsigned short*       hdst = hbuf + (size_t)((s & 1) ^ 1) * (B_ * H_);

    v8f acc[4];
#pragma unroll
    for (int g = 0; g < 4; ++g) acc[g] = vzero();

    for (int kc = 0; kc < 16; ++kc) {
      Frag a;
      const char* ap = (const char*)hsrc + (size_t)((row_a * H_ + kc * 32 + hs * 16) * 2);
      a.q[0] = *(const u32x4*)ap;
      a.q[1] = *(const u32x4*)(ap + 16);
#pragma unroll
      for (int g = 0; g < 4; ++g) {
        Frag b;
        const char* bp = (const char*)shB + (size_t)((((g * 2 + p) * 16 + kc) * 1024) + lane * 32);
        b.q[0] = ((const u32x4*)bp)[0];
        b.q[1] = ((const u32x4*)bp)[1];
        acc[g] = wmma_bf16(a.v, b.v, acc[g]);
      }
    }

    // Per-row partial LN stats over this wave's 64 gate columns.
    float ps[8], pq[8];
#pragma unroll
    for (int r = 0; r < 8; ++r) {
      float s1 = 0.f, q1 = 0.f;
#pragma unroll
      for (int g = 0; g < 4; ++g) { float v = acc[g][r]; s1 += v; q1 += v * v; }
      for (int mk = 1; mk < 16; mk <<= 1) {
        s1 += __shfl_xor(s1, mk, 32);
        q1 += __shfl_xor(q1, mk, 32);
      }
      ps[r] = s1; pq[r] = q1;
    }
    if (nlo == 0) {
#pragma unroll
      for (int r = 0; r < 8; ++r) {
        shSt[((m * 2 + p) * 16 + hs * 8 + r) * 2 + 0] = ps[r];
        shSt[((m * 2 + p) * 16 + hs * 8 + r) * 2 + 1] = pq[r];
      }
    }
    __syncthreads();
    if (tid < 64) {
      const int mr = tid >> 4, lr = tid & 15;
      float s0 = shSt[((mr * 2 + 0) * 16 + lr) * 2 + 0] + shSt[((mr * 2 + 1) * 16 + lr) * 2 + 0];
      float q0 = shSt[((mr * 2 + 0) * 16 + lr) * 2 + 1] + shSt[((mr * 2 + 1) * 16 + lr) * 2 + 1];
      gstats[(wg * 64 + tid) * 2 + 0] = s0;
      gstats[(wg * 64 + tid) * 2 + 1] = q0;
    }
    gridbar(bar, bexp);
    if (tid < 64) {
      float s0 = 0.f, q0 = 0.f;
      for (int w2 = 0; w2 < NWGR_; ++w2) {
        s0 += gstats[(w2 * 64 + tid) * 2 + 0];
        q0 += gstats[(w2 * 64 + tid) * 2 + 1];
      }
      float mu  = s0 * (1.0f / (float)G4_);
      float var = q0 * (1.0f / (float)G4_) - mu * mu;
      shMu[tid] = mu;
      shRs[tid] = rsqrtf(var + LN_EPS);
    }
    __syncthreads();

#pragma unroll
    for (int r = 0; r < 8; ++r) {
      const int row = crow[r];
      const float mu = shMu[row], rsv = shRs[row];
      float gt[4];
#pragma unroll
      for (int g = 0; g < 4; ++g) {
        const float ln  = (acc[g][r] - mu) * rsv * gg[g] + gb[g];
        const float gxv = b2f(gx[(size_t)(s * B_ + row) * G4_ + colg[g]]);
        gt[g] = ln + gxv + bia[g];
      }
      const float iv = sigm(gt[0]), fv = sigm(gt[1]), ov = sigm(gt[2]), zv = tanhf(gt[3]);
      cst[r] = fv * cst[r] + iv * zv;
      const float hv = ov * tanhf(cst[r]);
      hfin[r] = hv;
      hdst[row * H_ + kcol] = f2b(hv);
      if (is_last) outx[(size_t)(s * B_ + row) * H_ + kcol] = hv;
      else         x1b[(size_t)(s * B_ + row) * H_ + kcol] = f2b(hv);
    }
    gridbar(bar, bexp);
  }

#pragma unroll
  for (int r = 0; r < 8; ++r) {
    outh[crow[r] * H_ + kcol] = hfin[r];
    outc[crow[r] * H_ + kcol] = cst[r];
  }
}

// ---------------------------------------------------------------------------
extern "C" void kernel_launch(void* const* d_in, const int* in_sizes, int n_in,
                              void* d_out, int out_size, void* d_ws, size_t ws_size,
                              hipStream_t stream) {
  (void)in_sizes; (void)n_in; (void)out_size; (void)ws_size;
  const float* inputs = (const float*)d_in[0];
  const float* h0     = (const float*)d_in[1];
  const float* c0     = (const float*)d_in[2];
  const float* wx     = (const float*)d_in[3];
  const float* wh     = (const float*)d_in[4];
  const float* bias   = (const float*)d_in[5];
  const float* ln_gx  = (const float*)d_in[6];
  const float* ln_bx  = (const float*)d_in[7];
  const float* ln_gh  = (const float*)d_in[8];
  const float* ln_bh  = (const float*)d_in[9];

  char* w = (char*)d_ws;
  size_t cur = 0;
  auto alloc = [&](size_t bytes) -> char* {
    char* pp = w + cur;
    cur += (bytes + 255) & ~(size_t)255;
    return pp;
  };
  unsigned short* wxp  = (unsigned short*)alloc((size_t)2 * D_ * G4_ * 2);   //   4 MB
  unsigned short* whp  = (unsigned short*)alloc((size_t)2 * D_ * G4_ * 2);   //   4 MB
  unsigned short* xb   = (unsigned short*)alloc((size_t)NSB_ * D_ * 2);      //  32 MB
  unsigned short* x1b  = (unsigned short*)alloc((size_t)NSB_ * H_ * 2);      //  32 MB
  unsigned short* gxb  = (unsigned short*)alloc((size_t)NSB_ * G4_ * 2);     // 128 MB
  unsigned short* hbuf = (unsigned short*)alloc((size_t)2 * B_ * H_ * 2);    // ping-pong h
  float*          gst  = (float*)alloc((size_t)NWGR_ * 64 * 2 * 4);
  unsigned*       bar  = (unsigned*)alloc(256);

  float* out  = (float*)d_out;
  float* outx = out;                               // [S,B,H]
  float* outh = out + (size_t)NSB_ * H_;           // [L,B,H]
  float* outc = outh + (size_t)2 * B_ * H_;        // [L,B,H]

  const size_t REC_LDS = 131072 + 1024 + 512;
  hipFuncSetAttribute((const void*)k_rec, hipFuncAttributeMaxDynamicSharedMemorySize,
                      (int)REC_LDS);

  k_pack<<<dim3(8192), dim3(256), 0, stream>>>(wx, wh, wxp, whp);
  k_cvt<<<dim3((NSB_ * D_) / 256), dim3(256), 0, stream>>>(inputs, xb, NSB_ * D_);

  for (int l = 0; l < 2; ++l) {
    const unsigned short* xin = l ? x1b : xb;
    k_p1<<<dim3(NSB_ / 16), dim3(256), 0, stream>>>(
        xin, wxp + (size_t)l * D_ * G4_, ln_gx + l * G4_, ln_bx + l * G4_, gxb);
    k_initH<<<dim3(128), dim3(256), 0, stream>>>(h0 + (size_t)l * B_ * H_, hbuf);
    hipMemsetAsync(bar, 0, 8, stream);
    k_rec<<<dim3(NWGR_), dim3(256), REC_LDS, stream>>>(
        gxb, whp + (size_t)l * D_ * G4_, bias + l * G4_, ln_gh + l * G4_, ln_bh + l * G4_,
        c0 + (size_t)l * B_ * H_, hbuf, gst, bar,
        outx, x1b, outh + (size_t)l * B_ * H_, outc + (size_t)l * B_ * H_, l == 1);
  }
}
